// Inception_Conv_Attention_Block_1279900254214
// MI455X (gfx1250) — matrix operand order repaired and checked by reading, not compiled, not run on hardware
//
#include <hip/hip_runtime.h>
#include <math.h>

// ---------------------------------------------------------------------------
// Inception_Conv_Attention_Block for MI455X (gfx1250, wave32, WMMA)
// ---------------------------------------------------------------------------
#define B_   8
#define L_   2048
#define C_   512
#define CR_  128
#define DH_  256
#define M_   (B_ * L_)              // 16384 rows
#define EPS_LN 1e-5f

static const size_t SZ = (size_t)M_ * C_;   // 8,388,608 floats per (B,L,C) tensor

// workspace layout (in floats)
#define OFF_XP    ((size_t)0)                 // xp, later reused as "fused"
#define OFF_FEAT  (SZ)                        // 4 branch tensors
#define OFF_G3    (OFF_FEAT)                  // gemm3 out reuses feat0 (feat dead after fuse)
#define OFF_YM    (OFF_FEAT + 4 * SZ)
#define OFF_WT    (OFF_YM + (size_t)4 * B_ * C_)
#define OFF_QV    (OFF_WT + (size_t)4 * B_ * C_)
#define OFF_SC    (OFF_QV + DH_)
#define OFF_PWS   (OFF_SC + (size_t)M_ * 4)
#define OFF_PWT   (OFF_PWS + C_)
#define OFF_DWS   (OFF_PWT + C_)
#define OFF_DWT   (OFF_DWS + (size_t)4 * C_)

typedef __attribute__((ext_vector_type(16))) _Float16 v16h;
typedef __attribute__((ext_vector_type(8)))  float    v8f;

// ---------------------------------------------------------------------------
// WMMA fragment helpers (CDNA5 16-bit A 16x32, B 32x16, f32 C/D layouts)
// All per-lane runs are contiguous and 16B-aligned -> float4 (b128) loads.
// ---------------------------------------------------------------------------
// A (16xK row-major activations): lane L holds row m=L&15; halves 0..7 are
// K = k0 + kb + j, halves 8..15 are K = k0 + 16 + kb + j, kb = (L&16)?8:0.
__device__ __forceinline__ v16h load_a_frag(const float* __restrict__ A, int ldk,
                                            int m, int k0, int lane) {
  const int kb = (lane & 16) ? 8 : 0;
  const float4* p = (const float4*)(A + (size_t)m * ldk + k0 + kb);
  const float4 q0 = p[0];           // K+0..3
  const float4 q1 = p[1];           // K+4..7
  const float4 q2 = p[4];           // K+16..19
  const float4 q3 = p[5];           // K+20..23
  v16h a;
  a[0]  = (_Float16)q0.x; a[1]  = (_Float16)q0.y; a[2]  = (_Float16)q0.z; a[3]  = (_Float16)q0.w;
  a[4]  = (_Float16)q1.x; a[5]  = (_Float16)q1.y; a[6]  = (_Float16)q1.z; a[7]  = (_Float16)q1.w;
  a[8]  = (_Float16)q2.x; a[9]  = (_Float16)q2.y; a[10] = (_Float16)q2.z; a[11] = (_Float16)q2.w;
  a[12] = (_Float16)q3.x; a[13] = (_Float16)q3.y; a[14] = (_Float16)q3.z; a[15] = (_Float16)q3.w;
  return a;
}

// B is W^T where W is (N,K) row-major: B[k][n] = W[n*K + k]. Lane L holds
// col n=L&15; halves 0..7 are K = k0 + kb + j, halves 8..15 are
// K = k0 + 16 + kb + j, kb = (L&16)?8:0 (same order as A).
__device__ __forceinline__ v16h load_b_frag(const float* __restrict__ W, int ldk,
                                            int n, int k0, int lane) {
  const int kb = (lane & 16) ? 8 : 0;
  const float4* p = (const float4*)(W + (size_t)n * ldk + k0 + kb);
  const float4 q0 = p[0];
  const float4 q1 = p[1];
  const float4 q2 = p[4];
  const float4 q3 = p[5];
  v16h b;
  b[0]  = (_Float16)q0.x; b[1]  = (_Float16)q0.y; b[2]  = (_Float16)q0.z; b[3]  = (_Float16)q0.w;
  b[4]  = (_Float16)q1.x; b[5]  = (_Float16)q1.y; b[6]  = (_Float16)q1.z; b[7]  = (_Float16)q1.w;
  b[8]  = (_Float16)q2.x; b[9]  = (_Float16)q2.y; b[10] = (_Float16)q2.z; b[11] = (_Float16)q2.w;
  b[12] = (_Float16)q3.x; b[13] = (_Float16)q3.y; b[14] = (_Float16)q3.z; b[15] = (_Float16)q3.w;
  return b;
}

__device__ __forceinline__ v8f wmma_acc(v16h a, v16h b, v8f c) {
  // (neg_a, A, neg_b, B, c_mod, C, reuse_a, reuse_b)
  return __builtin_amdgcn_wmma_f32_16x16x32_f16(false, a, false, b, (short)0, c,
                                                false, false);
}

// ---------------------------------------------------------------------------
// Generic 512-K WMMA GEMM: O[M,N] = act(A[M,512] @ W[N,512]^T)
// MODE 0: BN+ReLU epilogue  (v*S[n]+T[n], relu)   -- pointwise conv
// MODE 1: bias epilogue     (v+S[n])              -- output projection
// grid: (M/128, N/64), block 256 (8 waves). Each wave: 16 rows x 64 cols.
// ---------------------------------------------------------------------------
template <int MODE>
__global__ void gemm512_kernel(const float* __restrict__ A, const float* __restrict__ W,
                               const float* __restrict__ S, const float* __restrict__ T,
                               float* __restrict__ O, int N) {
  const int lane = threadIdx.x & 31;
  const int wave = threadIdx.x >> 5;
  const int m0 = (blockIdx.x * 8 + wave) * 16;
  const int n0 = blockIdx.y * 64;
  const int mA = m0 + (lane & 15);

  v8f acc[4];
#pragma unroll
  for (int t = 0; t < 4; ++t)
#pragma unroll
    for (int r = 0; r < 8; ++r) acc[t][r] = 0.f;

#pragma unroll 4
  for (int k0 = 0; k0 < C_; k0 += 32) {
    if (k0 + 32 < C_)  // gfx1250 global_prefetch_b8 of next A chunk
      __builtin_prefetch(A + (size_t)mA * C_ + k0 + 32, 0, 1);
    const v16h a = load_a_frag(A, C_, mA, k0, lane);
#pragma unroll
    for (int t = 0; t < 4; ++t) {
      const v16h b = load_b_frag(W, C_, n0 + t * 16 + (lane & 15), k0, lane);
      acc[t] = wmma_acc(a, b, acc[t]);
    }
  }

  const int nlo = lane & 15;
  const int mb  = m0 + ((lane & 16) ? 8 : 0);
#pragma unroll
  for (int t = 0; t < 4; ++t) {
    const int n = n0 + t * 16 + nlo;
#pragma unroll
    for (int r = 0; r < 8; ++r) {
      float v = acc[t][r];
      if (MODE == 0) v = fmaxf(v * S[n] + T[n], 0.f);
      else           v = v + S[n];
      O[(size_t)(mb + r) * N + n] = v;
    }
  }
}

// ---------------------------------------------------------------------------
// Fused score kernel: score[m,br] = sum_d tanh(q[d]*(E@wv^T + bv)[m,d])*wg[d]+bg
// grid (M/128, 4 branches), block 256. Wave: 16 rows x full DH=256.
// K outermost: one A fragment feeds 16 back-to-back WMMAs (16 accumulators).
// ---------------------------------------------------------------------------
__global__ void score_kernel(const float* __restrict__ FEAT, const float* __restrict__ WV,
                             const float* __restrict__ BV, const float* __restrict__ QV,
                             const float* __restrict__ WG, const float* __restrict__ BG,
                             float* __restrict__ SC) {
  const int br = blockIdx.y;
  const float* E = FEAT + (size_t)br * SZ;
  const int lane = threadIdx.x & 31;
  const int wave = threadIdx.x >> 5;
  const int m0 = (blockIdx.x * 8 + wave) * 16;
  const int mA = m0 + (lane & 15);

  v8f acc[DH_ / 16];
#pragma unroll
  for (int nt = 0; nt < DH_ / 16; ++nt)
#pragma unroll
    for (int r = 0; r < 8; ++r) acc[nt][r] = 0.f;

  for (int k0 = 0; k0 < C_; k0 += 32) {
    if (k0 + 32 < C_)
      __builtin_prefetch(E + (size_t)mA * C_ + k0 + 32, 0, 1);
    const v16h a = load_a_frag(E, C_, mA, k0, lane);
#pragma unroll
    for (int nt = 0; nt < DH_ / 16; ++nt) {
      const v16h b = load_b_frag(WV, C_, nt * 16 + (lane & 15), k0, lane);
      acc[nt] = wmma_acc(a, b, acc[nt]);
    }
  }

  float sAcc[8];
#pragma unroll
  for (int r = 0; r < 8; ++r) sAcc[r] = 0.f;
#pragma unroll
  for (int nt = 0; nt < DH_ / 16; ++nt) {
    const int n = nt * 16 + (lane & 15);
    const float q = QV[n], bv = BV[n], g = WG[n];
#pragma unroll
    for (int r = 0; r < 8; ++r) sAcc[r] += tanhf(q * (acc[nt][r] + bv)) * g;
  }

  const float bg = BG[0];
#pragma unroll
  for (int r = 0; r < 8; ++r) {
    float v = sAcc[r];
    v += __shfl_xor(v, 1, 32);
    v += __shfl_xor(v, 2, 32);
    v += __shfl_xor(v, 4, 32);
    v += __shfl_xor(v, 8, 32);   // reduced within each 16-lane half
    const int m = m0 + r + ((lane & 16) ? 8 : 0);
    if ((lane & 15) == 0) SC[(size_t)m * 4 + br] = v + bg;
  }
}

// ---------------------------------------------------------------------------
// BN fold: s = g*rsqrt(v+eps), t = (bias - m)*s + b   (pointwise + 4 depthwise)
// ---------------------------------------------------------------------------
__global__ void prep_bn_kernel(const float* __restrict__ pwb, const float* __restrict__ pwg,
                               const float* __restrict__ pwbb, const float* __restrict__ pwm,
                               const float* __restrict__ pwv,
                               const float* __restrict__ dwb, const float* __restrict__ dwg,
                               const float* __restrict__ dwbb, const float* __restrict__ dwm,
                               const float* __restrict__ dwv,
                               float* __restrict__ PWS, float* __restrict__ PWT,
                               float* __restrict__ DWS, float* __restrict__ DWT) {
  const int c = threadIdx.x;
  float s = pwg[c] * rsqrtf(pwv[c] + EPS_LN);
  PWS[c] = s;
  PWT[c] = (pwb[c] - pwm[c]) * s + pwbb[c];
#pragma unroll
  for (int br = 0; br < 4; ++br) {
    const int i = br * C_ + c;
    float si = dwg[i] * rsqrtf(dwv[i] + EPS_LN);
    DWS[i] = si;
    DWT[i] = (dwb[i] - dwm[i]) * si + dwbb[i];
  }
}

// q[d] = bq[d] + sum_c df_ln_b[c] * wq[d,c]
// (LN of a channel-constant row is exactly its beta; Q is row-invariant)
__global__ void qvec_kernel(const float* __restrict__ DFB, const float* __restrict__ WQ,
                            const float* __restrict__ BQ, float* __restrict__ QV) {
  const int d = threadIdx.x;
  float a = BQ[d];
  for (int c = 0; c < C_; ++c) a += DFB[c] * WQ[(size_t)d * C_ + c];
  QV[d] = a;
}

// ---------------------------------------------------------------------------
// Depthwise conv along L (stride C_ layout), fused bias+BN+ReLU.
// branch 3 (k=20) has conv length L-1; position L-1 is zero-padded post-ReLU.
// ---------------------------------------------------------------------------
__global__ void dw_kernel(const float* __restrict__ XP,
                          const float* __restrict__ W0, const float* __restrict__ W1,
                          const float* __restrict__ W2, const float* __restrict__ W3,
                          const float* __restrict__ DWS, const float* __restrict__ DWT,
                          float* __restrict__ FEAT) {
  const int br = blockIdx.y;
  const size_t idx = (size_t)blockIdx.x * 256 + threadIdx.x;
  const int c = (int)(idx & (C_ - 1));
  const size_t m = idx >> 9;
  const int l = (int)(m & (L_ - 1));
  const size_t b = m >> 11;
  const int ks[4] = {3, 7, 15, 20};
  const int ps[4] = {1, 3, 7, 9};
  float out = 0.f;
  if (!(br == 3 && l == L_ - 1)) {
    const int k = ks[br], p = ps[br];
    const float* w = (br == 0 ? W0 : br == 1 ? W1 : br == 2 ? W2 : W3) + (size_t)c * k;
    const float* xc = XP + b * L_ * C_ + c;
    float acc = 0.f;
    for (int t = 0; t < k; ++t) {
      const int li = l + t - p;
      if (li >= 0 && li < L_) acc += xc[(size_t)li * C_] * w[t];
    }
    out = fmaxf(acc * DWS[br * C_ + c] + DWT[br * C_ + c], 0.f);
  }
  FEAT[(size_t)br * SZ + idx] = out;
}

// ym[br,b,c] = mean_l feat[br][b,l,c].  grid 32 (br*8+b), block 512.
__global__ void se_mean_kernel(const float* __restrict__ FEAT, float* __restrict__ YM) {
  const int br = blockIdx.x >> 3;
  const int b  = blockIdx.x & 7;
  const int c  = threadIdx.x;
  const float* f = FEAT + (size_t)br * SZ + (size_t)b * L_ * C_ + c;
  float s = 0.f;
  for (int l = 0; l < L_; ++l) s += f[(size_t)l * C_];
  YM[((size_t)br * B_ + b) * C_ + c] = s * (1.f / (float)L_);
}

// SE FC stack: h = relu(ym@w1^T+b1); wt = sigmoid(h@w2^T+b2). grid 32, block 256.
__global__ void se_fc_kernel(const float* __restrict__ YM,
                             const float* __restrict__ W1, const float* __restrict__ B1,
                             const float* __restrict__ W2, const float* __restrict__ B2,
                             float* __restrict__ WT) {
  __shared__ float ym[C_];
  __shared__ float h[CR_];
  const int br = blockIdx.x >> 3;
  const int b  = blockIdx.x & 7;
  const int t  = threadIdx.x;
  const float* y = YM + ((size_t)br * B_ + b) * C_;
  ym[t] = y[t];
  ym[t + 256] = y[t + 256];
  __syncthreads();
  if (t < CR_) {
    float a = B1[br * CR_ + t];
    const float* w = W1 + ((size_t)br * CR_ + t) * C_;
    for (int c = 0; c < C_; ++c) a += ym[c] * w[c];
    h[t] = fmaxf(a, 0.f);
  }
  __syncthreads();
  for (int c = t; c < C_; c += 256) {
    float a = B2[br * C_ + c];
    const float* w = W2 + ((size_t)br * C_ + c) * CR_;
    for (int j = 0; j < CR_; ++j) a += h[j] * w[j];
    WT[((size_t)br * B_ + b) * C_ + c] = 1.f / (1.f + expf(-a));
  }
}

// ---------------------------------------------------------------------------
// block-wide dual reduction (sum, sumsq), blockDim multiple of 32
// ---------------------------------------------------------------------------
__device__ __forceinline__ void block_reduce2(float& a, float& b, float* sm) {
  const int lane = threadIdx.x & 31, wid = threadIdx.x >> 5;
#pragma unroll
  for (int o = 16; o; o >>= 1) {
    a += __shfl_xor(a, o, 32);
    b += __shfl_xor(b, o, 32);
  }
  if (lane == 0) { sm[wid * 2] = a; sm[wid * 2 + 1] = b; }
  __syncthreads();
  if (threadIdx.x == 0) {
    float sa = 0.f, sb = 0.f;
    const int nw = blockDim.x >> 5;
    for (int w = 0; w < nw; ++w) { sa += sm[w * 2]; sb += sm[w * 2 + 1]; }
    sm[0] = sa; sm[1] = sb;
  }
  __syncthreads();
  a = sm[0];
  b = sm[1];
  __syncthreads();
}

// enhanced = LN(feat*(1+wt)) in place. grid (M_, 4), block 256.
__global__ void se_ln_kernel(float* __restrict__ FEAT, const float* __restrict__ WT,
                             const float* __restrict__ G, const float* __restrict__ Bb) {
  __shared__ float sm[32];
  const int br = blockIdx.y;
  const size_t row = blockIdx.x;
  const int batch = (int)(row >> 11);
  float* f = FEAT + (size_t)br * SZ + row * C_;
  const float* wt = WT + ((size_t)br * B_ + batch) * C_;
  const float* g  = G + (size_t)br * C_;
  const float* bb = Bb + (size_t)br * C_;
  const int t = threadIdx.x;
  const float v0 = f[t]       * (1.f + wt[t]);
  const float v1 = f[t + 256] * (1.f + wt[t + 256]);
  float s = v0 + v1, sq = v0 * v0 + v1 * v1;
  block_reduce2(s, sq, sm);
  const float mean = s * (1.f / (float)C_);
  const float inv  = rsqrtf(sq * (1.f / (float)C_) - mean * mean + EPS_LN);
  f[t]       = (v0 - mean) * inv * g[t] + bb[t];
  f[t + 256] = (v1 - mean) * inv * g[t + 256] + bb[t + 256];
}

// softmax over 4 branch scores + weighted fusion. grid M_*C_/256, block 256.
__global__ void fuse_kernel(const float* __restrict__ SC, const float* __restrict__ FEAT,
                            float* __restrict__ FUSED) {
  const size_t idx = (size_t)blockIdx.x * 256 + threadIdx.x;
  const size_t row = idx >> 9;
  const float s0 = SC[row * 4 + 0], s1 = SC[row * 4 + 1];
  const float s2 = SC[row * 4 + 2], s3 = SC[row * 4 + 3];
  const float mx = fmaxf(fmaxf(s0, s1), fmaxf(s2, s3));
  const float e0 = expf(s0 - mx), e1 = expf(s1 - mx);
  const float e2 = expf(s2 - mx), e3 = expf(s3 - mx);
  const float inv = 1.f / (e0 + e1 + e2 + e3);
  FUSED[idx] = (e0 * FEAT[idx] + e1 * FEAT[SZ + idx] +
                e2 * FEAT[2 * SZ + idx] + e3 * FEAT[3 * SZ + idx]) * inv;
}

// out = LN( LN(g3)*opg+opb + x )*fng+fnb. grid M_, block 256.
__global__ void final_ln_kernel(const float* __restrict__ G3, const float* __restrict__ X,
                                const float* __restrict__ OPG, const float* __restrict__ OPB,
                                const float* __restrict__ FNG, const float* __restrict__ FNB,
                                float* __restrict__ OUT) {
  __shared__ float sm[32];
  const size_t row = blockIdx.x;
  const int t = threadIdx.x;
  const float* g3 = G3 + row * C_;
  const float* x  = X + row * C_;
  const float t0 = g3[t], t1 = g3[t + 256];
  float s = t0 + t1, sq = t0 * t0 + t1 * t1;
  block_reduce2(s, sq, sm);
  float m1 = s * (1.f / (float)C_);
  float inv1 = rsqrtf(sq * (1.f / (float)C_) - m1 * m1 + EPS_LN);
  const float u0 = (t0 - m1) * inv1 * OPG[t] + OPB[t];
  const float u1 = (t1 - m1) * inv1 * OPG[t + 256] + OPB[t + 256];
  const float w0 = u0 + x[t];
  const float w1 = u1 + x[t + 256];
  s = w0 + w1; sq = w0 * w0 + w1 * w1;
  block_reduce2(s, sq, sm);
  float m2 = s * (1.f / (float)C_);
  float inv2 = rsqrtf(sq * (1.f / (float)C_) - m2 * m2 + EPS_LN);
  OUT[row * C_ + t]       = (w0 - m2) * inv2 * FNG[t] + FNB[t];
  OUT[row * C_ + t + 256] = (w1 - m2) * inv2 * FNG[t + 256] + FNB[t + 256];
}

// ---------------------------------------------------------------------------
extern "C" void kernel_launch(void* const* d_in, const int* in_sizes, int n_in,
                              void* d_out, int out_size, void* d_ws, size_t ws_size,
                              hipStream_t stream) {
  const float* x       = (const float*)d_in[0];
  const float* pw_w    = (const float*)d_in[1];
  const float* pw_b    = (const float*)d_in[2];
  const float* pw_bn_g = (const float*)d_in[3];
  const float* pw_bn_b = (const float*)d_in[4];
  const float* pw_bn_m = (const float*)d_in[5];
  const float* pw_bn_v = (const float*)d_in[6];
  const float* dw_w0   = (const float*)d_in[7];
  const float* dw_w1   = (const float*)d_in[8];
  const float* dw_w2   = (const float*)d_in[9];
  const float* dw_w3   = (const float*)d_in[10];
  const float* dw_b    = (const float*)d_in[11];
  const float* dw_bn_g = (const float*)d_in[12];
  const float* dw_bn_b = (const float*)d_in[13];
  const float* dw_bn_m = (const float*)d_in[14];
  const float* dw_bn_v = (const float*)d_in[15];
  const float* se_w1   = (const float*)d_in[16];
  const float* se_b1   = (const float*)d_in[17];
  const float* se_w2   = (const float*)d_in[18];
  const float* se_b2   = (const float*)d_in[19];
  const float* se_ln_g = (const float*)d_in[20];
  const float* se_ln_b = (const float*)d_in[21];
  // d_in[22] = df_ln_g (multiplies an exact zero; unused)
  const float* df_ln_b = (const float*)d_in[23];
  const float* wq      = (const float*)d_in[24];
  const float* bq      = (const float*)d_in[25];
  const float* wv      = (const float*)d_in[26];
  const float* bv      = (const float*)d_in[27];
  const float* wg      = (const float*)d_in[28];
  const float* bg      = (const float*)d_in[29];
  const float* op_w    = (const float*)d_in[30];
  const float* op_b    = (const float*)d_in[31];
  const float* op_ln_g = (const float*)d_in[32];
  const float* op_ln_b = (const float*)d_in[33];
  const float* fn_g    = (const float*)d_in[34];
  const float* fn_b    = (const float*)d_in[35];

  float* ws   = (float*)d_ws;
  float* XP   = ws + OFF_XP;     // also FUSED
  float* FEAT = ws + OFF_FEAT;   // 4 branches; feat0 later reused as G3
  float* G3   = ws + OFF_G3;
  float* YM   = ws + OFF_YM;
  float* WT   = ws + OFF_WT;
  float* QV   = ws + OFF_QV;
  float* SCp  = ws + OFF_SC;
  float* PWS  = ws + OFF_PWS;
  float* PWT  = ws + OFF_PWT;
  float* DWS  = ws + OFF_DWS;
  float* DWT  = ws + OFF_DWT;

  // 1. fold BN params, compute constant Q vector
  prep_bn_kernel<<<1, 512, 0, stream>>>(pw_b, pw_bn_g, pw_bn_b, pw_bn_m, pw_bn_v,
                                        dw_b, dw_bn_g, dw_bn_b, dw_bn_m, dw_bn_v,
                                        PWS, PWT, DWS, DWT);
  qvec_kernel<<<1, DH_, 0, stream>>>(df_ln_b, wq, bq, QV);

  // 2. pointwise conv (WMMA GEMM, BN+ReLU epilogue): xp[m,n]
  gemm512_kernel<0><<<dim3(M_ / 128, C_ / 64), 256, 0, stream>>>(x, pw_w, PWS, PWT, XP, C_);

  // 3. four depthwise convs + bias/BN/ReLU -> feat[br]
  dw_kernel<<<dim3((unsigned)(SZ / 256), 4), 256, 0, stream>>>(XP, dw_w0, dw_w1, dw_w2,
                                                               dw_w3, DWS, DWT, FEAT);

  // 4. SE gate
  se_mean_kernel<<<32, 512, 0, stream>>>(FEAT, YM);
  se_fc_kernel<<<32, 256, 0, stream>>>(YM, se_w1, se_b1, se_w2, se_b2, WT);

  // 5. enhanced = LN(feat*(1+wt)) in place
  se_ln_kernel<<<dim3(M_, 4), 256, 0, stream>>>(FEAT, WT, se_ln_g, se_ln_b);

  // 6. per-branch attention scores (fused WMMA GEMM + tanh reduction)
  score_kernel<<<dim3(M_ / 128, 4), 256, 0, stream>>>(FEAT, wv, bv, QV, wg, bg, SCp);

  // 7. softmax over branches + fusion -> FUSED (reuses XP region)
  fuse_kernel<<<(unsigned)(SZ / 256), 256, 0, stream>>>(SCp, FEAT, XP);

  // 8. output projection (WMMA GEMM + bias) -> G3 (reuses feat0 region)
  gemm512_kernel<1><<<dim3(M_ / 128, C_ / 64), 256, 0, stream>>>(XP, op_w, op_b, nullptr,
                                                                 G3, C_);

  // 9. LN -> +x residual -> LN -> out
  final_ln_kernel<<<M_, 256, 0, stream>>>(G3, x, op_ln_g, op_ln_b, fn_g, fn_b,
                                          (float*)d_out);
}
